// LSTMAutoEncoder_9148280340842
// MI455X (gfx1250) — matrix operand order repaired and checked by reading, not compiled, not run on hardware
//
#include <hip/hip_runtime.h>
#include <cstddef>

// ---------------------------------------------------------------------------
// LSTM autoencoder on gfx1250 (MI455X), wave32 + V_WMMA_F32_16X16X32_F16.
//
// One wave handles a 16-batch tile end-to-end (encoder 16 steps -> latent ->
// decoder 16 steps). Orientation: gates^T = W[16x32] x h^T[32x16batch], so
// A operand = weight rows (gate idx), B operand = hidden state (batch on
// lanes). The D/C layout then has batch on lanes -> LSTM nonlinearity is
// per-lane VALU; feeding h back as next B operand needs only a half-wave
// xor-16 exchange (ds_swizzle SWAPX16) + f32->f16 pack.
//
// Round-3 changes:
//  * Encoder x-projection WMMAs software-pipelined one timestep ahead: the
//    recurrent WMMAs consume xi produced a full iteration earlier, so the
//    scheduler cannot re-pair producer->consumer and no D->C hazard nops
//    are needed. The 8 xi WMMAs co-execute with the gate VALU math.
//  * x loads pipelined one step ahead; B_x zeros + bias slot built once
//    outside the loop (only x0/x1 refreshed per step, via cndmask not exec).
// ---------------------------------------------------------------------------

#define BATCH_N   131072
#define SEQ_T     16
#define HID       32

typedef __attribute__((ext_vector_type(16))) _Float16 v16h;
typedef __attribute__((ext_vector_type(8)))  float    v8f;

union U16h { v16h v; _Float16 h[16]; };
union U8f  { v8f  v; float f[8]; };

#define WMMA_F16(A, B, C) \
  __builtin_amdgcn_wmma_f32_16x16x32_f16(false, (A), false, (B), (short)0, (C), false, false)

// SWAPX16: ds_swizzle group-of-32, xor=0x10, and=0x1f  -> lane ^ 16
__device__ __forceinline__ float swz16(float v) {
  return __int_as_float(__builtin_amdgcn_ds_swizzle(__float_as_int(v), 0x401F));
}

// ---- activations: native v_tanh_f32 when available ------------------------
__device__ __forceinline__ float tanh_fast(float x) {
#if __has_builtin(__builtin_amdgcn_tanhf)
  return __builtin_amdgcn_tanhf(x);
#else
  float e = __expf(2.0f * x);
  return 1.0f - 2.0f * __builtin_amdgcn_rcpf(e + 1.0f);
#endif
}
__device__ __forceinline__ float sigf(float x) {
#if __has_builtin(__builtin_amdgcn_tanhf)
  return __builtin_amdgcn_tanhf(x * 0.5f) * 0.5f + 0.5f;
#else
  return __builtin_amdgcn_rcpf(1.0f + __expf(-x));
#endif
}

// Load one row of a [nrows x 32] f32 matrix into WMMA A-layout (f16).
// A operand, contiguous k order (same as B): lane<16 holds K = 0..15; lane>=16 holds
// K = 16..31; element e -> K = e + 16*hl.
__device__ __forceinline__ v16h load_a_row32(const float* __restrict__ W,
                                             int row, int nrows, int hl) {
  U16h u;
#pragma unroll
  for (int e = 0; e < 16; ++e) u.h[e] = (_Float16)0.0f;
  if (row < nrows) {
    const float* p = W + row * 32;
#pragma unroll
    for (int e = 0; e < 16; ++e) {
      int k = e + 16 * hl;
      u.h[e] = (_Float16)p[k];
    }
  }
  return u.v;
}

// Convert h (two 16x16 f32 C-layout tiles: rows = h-index, cols = batch on
// lanes) into the B operand in contiguous k order (lane<16: K=0..15, lane>=16: K=16..31).
__device__ __forceinline__ v16h build_Bh(v8f h0v, v8f h1v, int hl) {
  U8f h0, h1; h0.v = h0v; h1.v = h1v;
  U16h b;
#pragma unroll
  for (int v = 0; v < 8; ++v) {
    float s0 = swz16(h0.f[v]);          // partner half's tile-0 value
    float s1 = swz16(h1.f[v]);          // partner half's tile-1 value
    float lo = hl ? s1 : h0.f[v];       // K = v      (hl=0) | K = 16+v (hl=1)
    float hi = hl ? h1.f[v] : s0;       // K = 8+v    (hl=0) | K = 24+v (hl=1)
    b.h[v]     = (_Float16)lo;
    b.h[8 + v] = (_Float16)hi;
  }
  return b.v;
}

__device__ __forceinline__ void lstm_elem(v8f ai, v8f af, v8f ag, v8f ao,
                                          v8f& cs, v8f& hs) {
  U8f I, F, G, O, C, H;
  I.v = ai; F.v = af; G.v = ag; O.v = ao; C.v = cs;
#pragma unroll
  for (int v = 0; v < 8; ++v) {
    float cc = sigf(F.f[v]) * C.f[v] + sigf(I.f[v]) * tanh_fast(G.f[v]);
    C.f[v] = cc;
    H.f[v] = sigf(O.f[v]) * tanh_fast(cc);
  }
  cs = C.v; hs = H.v;
}

__global__ __launch_bounds__(256, 1)
void lstm_ae_kernel(const float* __restrict__ x,
                    const float* __restrict__ eWih, const float* __restrict__ eWhh,
                    const float* __restrict__ eBih, const float* __restrict__ eBhh,
                    const float* __restrict__ eFcW, const float* __restrict__ eFcB,
                    const float* __restrict__ dFcW, const float* __restrict__ dFcB,
                    const float* __restrict__ dWih, const float* __restrict__ dWhh,
                    const float* __restrict__ dBih, const float* __restrict__ dBhh,
                    const float* __restrict__ dOutW, const float* __restrict__ dOutB,
                    float* __restrict__ out) {
  const int lane = threadIdx.x & 31;
  const int hl   = lane >> 4;            // half-wave: 0 -> lanes 0..15
  const int col  = lane & 15;            // batch column within tile / A row
  const int tile = (int)((blockIdx.x * blockDim.x + threadIdx.x) >> 5);
  const int b    = tile * 16 + col;      // batch element owned by this lane

  const v8f z8 = {0.f, 0.f, 0.f, 0.f, 0.f, 0.f, 0.f, 0.f};
  const float fzero = 0.0f;

  __builtin_prefetch(x + (size_t)(tile * 16) * (SEQ_T * 2), 0, 0);

  // ---------------- encoder weights in registers -------------------------
  // gate chunk c covers gate rows 16c..16c+15; i:{0,1} f:{2,3} g:{4,5} o:{6,7}
  v16h Awhh[8], Awih[8];
#pragma unroll
  for (int c = 0; c < 8; ++c) {
    const int g = 16 * c + col;
    Awhh[c] = load_a_row32(eWhh, g, 128, hl);
    U16h u;
#pragma unroll
    for (int e = 0; e < 16; ++e) u.h[e] = (_Float16)0.0f;
    if (hl == 0) {                       // K slots 0,1 = W_ih; K slot 2 = bias
      u.h[0] = (_Float16)eWih[g * 2 + 0];
      u.h[1] = (_Float16)eWih[g * 2 + 1];
      u.h[2] = (_Float16)(eBih[g] + eBhh[g]);
    }
    Awih[c] = u.v;
  }

  v8f cs0 = z8, cs1 = z8, hs0 = z8, hs1 = z8;
  U16h bh0;
#pragma unroll
  for (int e = 0; e < 16; ++e) bh0.h[e] = (_Float16)0.0f;
  v16h Bh = bh0.v;                       // initial hidden = 0

  // B_x operand: zeros + constant K=2 bias slot built once; per step only
  // K=0 (x0) and K=1 (x1) are refreshed (cndmask keeps EXEC full).
  U16h bx;
#pragma unroll
  for (int e = 0; e < 16; ++e) bx.h[e] = (_Float16)0.0f;
  bx.h[2] = hl ? (_Float16)0.0f : (_Float16)1.0f;

  // ---------------- encoder time loop (xi pipelined by 1 step) ------------
  {
    const float* xp0 = x + ((size_t)b * SEQ_T + 0) * 2;
    float nx0 = xp0[0], nx1 = xp0[1];    // all lanes load (dup addr, cached)
    bx.h[0] = hl ? (_Float16)fzero : (_Float16)nx0;
    bx.h[1] = hl ? (_Float16)fzero : (_Float16)nx1;
  }
  v8f xi[8];
#pragma unroll
  for (int c = 0; c < 8; ++c) xi[c] = WMMA_F16(Awih[c], bx.v, z8);

  for (int t = 0; t < SEQ_T; ++t) {
    // recurrent WMMAs consume xi produced a full iteration earlier
    v8f ai0 = WMMA_F16(Awhh[0], Bh, xi[0]);
    v8f ai1 = WMMA_F16(Awhh[1], Bh, xi[1]);
    v8f af0 = WMMA_F16(Awhh[2], Bh, xi[2]);
    v8f af1 = WMMA_F16(Awhh[3], Bh, xi[3]);
    v8f ag0 = WMMA_F16(Awhh[4], Bh, xi[4]);
    v8f ag1 = WMMA_F16(Awhh[5], Bh, xi[5]);
    v8f ao0 = WMMA_F16(Awhh[6], Bh, xi[6]);
    v8f ao1 = WMMA_F16(Awhh[7], Bh, xi[7]);

    // pipeline: load x for t+1 and issue next x-projection WMMAs (these
    // co-execute with the gate VALU math below; last iteration's extras
    // are computed but unused -> loop body stays uniform, no divergence)
    {
      const int tn = (t + 1 < SEQ_T) ? (t + 1) : (SEQ_T - 1);
      const float* xp = x + ((size_t)b * SEQ_T + tn) * 2;
      float nx0 = xp[0], nx1 = xp[1];
      bx.h[0] = hl ? (_Float16)fzero : (_Float16)nx0;
      bx.h[1] = hl ? (_Float16)fzero : (_Float16)nx1;
    }
#pragma unroll
    for (int c = 0; c < 8; ++c) xi[c] = WMMA_F16(Awih[c], bx.v, z8);

    lstm_elem(ai0, af0, ag0, ao0, cs0, hs0);
    lstm_elem(ai1, af1, ag1, ao1, cs1, hs1);

    Bh = build_Bh(hs0, hs1, hl);         // h_t -> B operand for next step
  }
  // Bh now holds h_T in B layout.

  // ---------------- latent: z = enc_fc(h_T) ------------------------------
  v16h Az = load_a_row32(eFcW, col, 8, hl);   // rows 8..15 zero-padded
  U8f zb;
#pragma unroll
  for (int v = 0; v < 8; ++v) zb.f[v] = (hl == 0) ? eFcB[v] : 0.0f;
  v8f zt = WMMA_F16(Az, Bh, zb.v);            // z rows 0..7, batch on lanes

  // z (C layout) -> B operand: only K=0..7 nonzero, all in lanes<16.
  U16h bz;
#pragma unroll
  for (int e = 0; e < 16; ++e) bz.h[e] = (_Float16)0.0f;
  {
    U8f Z; Z.v = zt;
#pragma unroll
    for (int v = 0; v < 8; ++v) bz.h[v] = hl ? (_Float16)0.0f : (_Float16)Z.f[v];
  }

  // ---------------- h0 = dec_fc(z) ---------------------------------------
  v8f h0a = z8, h0b = z8;
#pragma unroll
  for (int c2 = 0; c2 < 2; ++c2) {
    U16h u;
#pragma unroll
    for (int e = 0; e < 16; ++e) u.h[e] = (_Float16)0.0f;
    if (hl == 0) {                       // K=0..7 = dec_fc_W row (L=8 inputs)
      const float* p = dFcW + (16 * c2 + col) * 8;
#pragma unroll
      for (int e = 0; e < 8; ++e) u.h[e] = (_Float16)p[e];
    }
    U8f hb;
#pragma unroll
    for (int v = 0; v < 8; ++v) hb.f[v] = dFcB[16 * c2 + 8 * hl + v];
    v8f r = WMMA_F16(u.v, bz.v, hb.v);
    if (c2 == 0) h0a = r; else h0b = r;
  }
  v16h Bh0 = build_Bh(h0a, h0b, hl);     // decoder initial hidden = h0
  Bh = Bh0;

  // ---------------- decoder constants ------------------------------------
  // x_proj = W_ih * h0 + (b_ih + b_hh): constant over time, compute once.
  v8f xproj[8];
#pragma unroll
  for (int c = 0; c < 8; ++c) {
    const int g0 = 16 * c + 8 * hl;
    U8f cb;
#pragma unroll
    for (int v = 0; v < 8; ++v) cb.f[v] = dBih[g0 + v] + dBhh[g0 + v];
    v16h Aw = load_a_row32(dWih, 16 * c + col, 128, hl);
    xproj[c] = WMMA_F16(Aw, Bh0, cb.v);
  }
  v16h Dwhh[8];
#pragma unroll
  for (int c = 0; c < 8; ++c) Dwhh[c] = load_a_row32(dWhh, 16 * c + col, 128, hl);

  v16h Aout = load_a_row32(dOutW, col, 2, hl);   // rows 2..15 zero
  U8f ob;
#pragma unroll
  for (int v = 0; v < 8; ++v) ob.f[v] = (hl == 0 && v < 2) ? dOutB[v] : 0.0f;

  cs0 = z8; cs1 = z8;                    // decoder c0 = 0, h0 already in Bh

  // ---------------- decoder time loop ------------------------------------
  for (int t = 0; t < SEQ_T; ++t) {
    v8f ai0 = WMMA_F16(Dwhh[0], Bh, xproj[0]);
    v8f ai1 = WMMA_F16(Dwhh[1], Bh, xproj[1]);
    v8f af0 = WMMA_F16(Dwhh[2], Bh, xproj[2]);
    v8f af1 = WMMA_F16(Dwhh[3], Bh, xproj[3]);
    v8f ag0 = WMMA_F16(Dwhh[4], Bh, xproj[4]);
    v8f ag1 = WMMA_F16(Dwhh[5], Bh, xproj[5]);
    v8f ao0 = WMMA_F16(Dwhh[6], Bh, xproj[6]);
    v8f ao1 = WMMA_F16(Dwhh[7], Bh, xproj[7]);

    lstm_elem(ai0, af0, ag0, ao0, cs0, hs0);
    lstm_elem(ai1, af1, ag1, ao1, cs1, hs1);

    Bh = build_Bh(hs0, hs1, hl);
    v8f ot = WMMA_F16(Aout, Bh, ob.v);   // rows 0,1 = output dims
    if (hl == 0) {
      U8f O; O.v = ot;
      float2 st; st.x = O.f[0]; st.y = O.f[1];
      *reinterpret_cast<float2*>(out + ((size_t)b * SEQ_T + t) * 2) = st;
    }
  }
}

extern "C" void kernel_launch(void* const* d_in, const int* in_sizes, int n_in,
                              void* d_out, int out_size, void* d_ws, size_t ws_size,
                              hipStream_t stream) {
  (void)in_sizes; (void)n_in; (void)out_size; (void)d_ws; (void)ws_size;
  const float* x     = (const float*)d_in[0];
  const float* eWih  = (const float*)d_in[1];
  const float* eWhh  = (const float*)d_in[2];
  const float* eBih  = (const float*)d_in[3];
  const float* eBhh  = (const float*)d_in[4];
  const float* eFcW  = (const float*)d_in[5];
  const float* eFcB  = (const float*)d_in[6];
  const float* dFcW  = (const float*)d_in[7];
  const float* dFcB  = (const float*)d_in[8];
  const float* dWih  = (const float*)d_in[9];
  const float* dWhh  = (const float*)d_in[10];
  const float* dBih  = (const float*)d_in[11];
  const float* dBhh  = (const float*)d_in[12];
  const float* dOutW = (const float*)d_in[13];
  const float* dOutB = (const float*)d_in[14];
  float* out = (float*)d_out;

  // 131072 batch / 16 per wave = 8192 waves; 8 waves (256 thr) per block.
  const int n_tiles = BATCH_N / 16;
  dim3 block(256);
  dim3 grid(n_tiles / 8);
  hipLaunchKernelGGL(lstm_ae_kernel, grid, block, 0, stream,
                     x, eWih, eWhh, eBih, eBhh, eFcW, eFcB,
                     dFcW, dFcB, dWih, dWhh, dBih, dBhh, dOutW, dOutB, out);
}
